// KLAttention2_33964601377283
// MI455X (gfx1250) — matrix operand order repaired and checked by reading, not compiled, not run on hardware
//
#include <hip/hip_runtime.h>
#include <hip/hip_bf16.h>

// Problem constants (reference: B=32, N=1024, D=1024, fp32 in/out)
constexpr int BB = 32;
constexpr int NN = 1024;
constexpr int DD = 1024;

typedef __bf16 v8bf  __attribute__((ext_vector_type(8)));
typedef __bf16 v16bf __attribute__((ext_vector_type(16)));
typedef float  v8f   __attribute__((ext_vector_type(8)));

#define SHUF16(lo, hi) __builtin_shufflevector((lo), (hi), 0,1,2,3,4,5,6,7,8,9,10,11,12,13,14,15)
#define WMMA_BF16(A, Bf, Cf) \
    __builtin_amdgcn_wmma_f32_16x16x32_bf16(false, (A), false, (Bf), (short)0, (Cf), false, false)

// ---------------------------------------------------------------------------
// Kernel 1: per-row softmax stats over D.
//   p[b,n,:]  = softmax(x[b,n,:])           (bf16)
//   xb        = bf16 copy of x
//   c[b,n]    = rowmax + log(sum exp)       (fp32)
// One wave per row; 8 rows per 256-thread block.
// ---------------------------------------------------------------------------
__global__ void k_row_stats(const float* __restrict__ X,
                            __bf16* __restrict__ P,
                            __bf16* __restrict__ XB,
                            float* __restrict__ C) {
    const int lane = threadIdx.x & 31;
    const int wave = threadIdx.x >> 5;
    const long row = (long)blockIdx.x * 8 + wave;
    const float* xr = X + row * DD;

    float v[DD / 32];
    float m = -3.0e38f;
#pragma unroll
    for (int k = 0; k < DD / 32; ++k) {
        float t = xr[lane + 32 * k];
        v[k] = t;
        m = fmaxf(m, t);
    }
#pragma unroll
    for (int off = 16; off > 0; off >>= 1) m = fmaxf(m, __shfl_xor(m, off, 32));

    float s = 0.0f;
#pragma unroll
    for (int k = 0; k < DD / 32; ++k) {
        v[k] = __expf(v[k] - m);
        s += v[k];
    }
#pragma unroll
    for (int off = 16; off > 0; off >>= 1) s += __shfl_xor(s, off, 32);

    const float inv  = 1.0f / s;
    const float logZ = __logf(s);

    __bf16* pr = P + row * DD;
    __bf16* xo = XB + row * DD;
#pragma unroll
    for (int k = 0; k < DD / 32; ++k) {
        const int idx = lane + 32 * k;
        pr[idx] = (__bf16)(v[k] * inv);
        xo[idx] = (__bf16)xr[idx];
    }
    if (lane == 0) C[row] = m + logZ;
}

// ---------------------------------------------------------------------------
// Kernel 2: XBT[b,d,n] = XB[b,n,d]  (bf16 transpose, per batch)
// 64x64 tile via LDS with pad-66 stride: conflict-free scalar LDS on both
// phases, fully coalesced 32B-vector global accesses on both phases.
// ---------------------------------------------------------------------------
__global__ void k_transpose(const __bf16* __restrict__ XB,
                            __bf16* __restrict__ XBT) {
    __shared__ __bf16 t[64 * 66];
    const int b  = blockIdx.z;
    const int n0 = blockIdx.x * 64;
    const int d0 = blockIdx.y * 64;
    const int r  = threadIdx.x >> 2;          // 0..63
    const int cb = (threadIdx.x & 3) * 16;    // 0,16,32,48

    // phase 1: read rows (n) of XB, scatter into padded LDS
    v16bf vin = *(const v16bf*)(XB + ((long)b * NN + n0 + r) * DD + d0 + cb);
#pragma unroll
    for (int q = 0; q < 16; ++q) t[r * 66 + cb + q] = vin[q];
    __syncthreads();

    // phase 2: gather columns of LDS, write rows (d) of XBT
    v16bf vout;
#pragma unroll
    for (int q = 0; q < 16; ++q) vout[q] = t[(cb + q) * 66 + r];
    *(v16bf*)(XBT + ((long)b * DD + d0 + r) * NN + n0 + cb) = vout;
}

// ---------------------------------------------------------------------------
// Kernel 3: S[b,i,j] = (p[b] @ xb[b]^T)[i,j] - c[b,j]     (fp32)
// Wave tile 32(i) x 64(j): 2 A-frags x 4 B-frags -> 8 WMMA per k-step,
// 12 b128 loads per 8 WMMA. Block = 8 waves = 64(i) x 256(j).
// ---------------------------------------------------------------------------
__global__ void k_gemm1(const __bf16* __restrict__ P,
                        const __bf16* __restrict__ XB,
                        const float* __restrict__ C,
                        float* __restrict__ S) {
    const int lane = threadIdx.x & 31;
    const int wave = threadIdx.x >> 5;
    const int b    = blockIdx.z;
    const int half = lane >> 4;
    const int nl   = lane & 15;
    const int i0   = (blockIdx.x * 2 + (wave & 1)) * 32;
    const int j0   = (blockIdx.y * 4 + (wave >> 1)) * 64;

    const __bf16* pA0 = P + ((long)b * NN + i0 + nl) * DD;
    const __bf16* pA1 = pA0 + 16 * DD;
    const __bf16* pB[4];
#pragma unroll
    for (int g = 0; g < 4; ++g)
        pB[g] = XB + ((long)b * NN + j0 + 16 * g + nl) * DD + 16 * half;

    v8f acc[2][4] = {};
    for (int kk = 0; kk < DD; kk += 32) {
        v16bf A0 = *(const v16bf*)(pA0 + kk + 16 * half);
        v16bf A1 = *(const v16bf*)(pA1 + kk + 16 * half);
#pragma unroll
        for (int g = 0; g < 4; ++g) {
            v16bf Bg = *(const v16bf*)(pB[g] + kk);
            acc[0][g] = WMMA_BF16(A0, Bg, acc[0][g]);
            acc[1][g] = WMMA_BF16(A1, Bg, acc[1][g]);
        }
    }

#pragma unroll
    for (int g = 0; g < 4; ++g) {
        const float cg = C[(long)b * NN + j0 + 16 * g + nl];
#pragma unroll
        for (int f = 0; f < 2; ++f) {
            float* Sout = S + ((long)b * NN + i0 + 16 * f + 8 * half) * NN + j0 + 16 * g + nl;
#pragma unroll
            for (int r = 0; r < 8; ++r) Sout[(long)r * NN] = acc[f][g][r] - cg;
        }
    }
}

// ---------------------------------------------------------------------------
// Kernel 4: attn[b,i,:] = softmax(S[b,i,:]) over j, stored bf16.
// ---------------------------------------------------------------------------
__global__ void k_attn_softmax(const float* __restrict__ S,
                               __bf16* __restrict__ A) {
    const int lane = threadIdx.x & 31;
    const int wave = threadIdx.x >> 5;
    const long row = (long)blockIdx.x * 8 + wave;
    const float* sr = S + row * NN;

    float v[NN / 32];
    float m = -3.0e38f;
#pragma unroll
    for (int k = 0; k < NN / 32; ++k) {
        float t = sr[lane + 32 * k];
        v[k] = t;
        m = fmaxf(m, t);
    }
#pragma unroll
    for (int off = 16; off > 0; off >>= 1) m = fmaxf(m, __shfl_xor(m, off, 32));

    float s = 0.0f;
#pragma unroll
    for (int k = 0; k < NN / 32; ++k) {
        v[k] = __expf(v[k] - m);
        s += v[k];
    }
#pragma unroll
    for (int off = 16; off > 0; off >>= 1) s += __shfl_xor(s, off, 32);
    const float inv = 1.0f / s;

    __bf16* ar = A + row * NN;
#pragma unroll
    for (int k = 0; k < NN / 32; ++k) ar[lane + 32 * k] = (__bf16)(v[k] * inv);
}

// ---------------------------------------------------------------------------
// Kernel 5: out[b] = attn[b] @ xb[b]   (fp32 out)
// Same structure as k_gemm1: B operand columns come from XBT rows, so all
// fragment loads are contiguous b128 -- no LDS, no barriers in the hot loop.
// ---------------------------------------------------------------------------
__global__ void k_gemm2(const __bf16* __restrict__ ATT,
                        const __bf16* __restrict__ XBT,
                        float* __restrict__ O) {
    const int lane = threadIdx.x & 31;
    const int wave = threadIdx.x >> 5;
    const int b    = blockIdx.z;
    const int half = lane >> 4;
    const int nl   = lane & 15;
    const int i0   = (blockIdx.x * 2 + (wave & 1)) * 32;
    const int d0   = (blockIdx.y * 4 + (wave >> 1)) * 64;

    const __bf16* pA0 = ATT + ((long)b * NN + i0 + nl) * NN;
    const __bf16* pA1 = pA0 + 16 * NN;
    const __bf16* pB[4];
#pragma unroll
    for (int g = 0; g < 4; ++g)
        pB[g] = XBT + ((long)b * DD + d0 + 16 * g + nl) * NN + 16 * half;

    v8f acc[2][4] = {};
    for (int kk = 0; kk < NN; kk += 32) {
        v16bf A0 = *(const v16bf*)(pA0 + kk + 16 * half);
        v16bf A1 = *(const v16bf*)(pA1 + kk + 16 * half);
#pragma unroll
        for (int g = 0; g < 4; ++g) {
            v16bf Bg = *(const v16bf*)(pB[g] + kk);
            acc[0][g] = WMMA_BF16(A0, Bg, acc[0][g]);
            acc[1][g] = WMMA_BF16(A1, Bg, acc[1][g]);
        }
    }

#pragma unroll
    for (int g = 0; g < 4; ++g) {
#pragma unroll
        for (int f = 0; f < 2; ++f) {
            float* out = O + ((long)b * NN + i0 + 16 * f + 8 * half) * DD + d0 + 16 * g + nl;
#pragma unroll
            for (int r = 0; r < 8; ++r) out[(long)r * DD] = acc[f][g][r];
        }
    }
}

// ---------------------------------------------------------------------------
// Launch
// ---------------------------------------------------------------------------
extern "C" void kernel_launch(void* const* d_in, const int* in_sizes, int n_in,
                              void* d_out, int out_size, void* d_ws, size_t ws_size,
                              hipStream_t stream) {
    const float* x = (const float*)d_in[0];
    float* out = (float*)d_out;

    // Workspace layout (bytes):
    //   P    bf16 [B,N,D]   64 MB
    //   XB   bf16 [B,N,D]   64 MB
    //   XBT  bf16 [B,D,N]   64 MB
    //   S    f32  [B,N,N]  128 MB
    //   ATT  bf16 [B,N,N]   64 MB
    //   C    f32  [B,N]    128 KB
    const size_t szPX = (size_t)BB * NN * DD * sizeof(__bf16);
    const size_t szS  = (size_t)BB * NN * NN * sizeof(float);
    const size_t szA  = (size_t)BB * NN * NN * sizeof(__bf16);
    char* w = (char*)d_ws;
    __bf16* P   = (__bf16*)(w);
    __bf16* XB  = (__bf16*)(w + szPX);
    __bf16* XBT = (__bf16*)(w + 2 * szPX);
    float*  S   = (float*)(w + 3 * szPX);
    __bf16* ATT = (__bf16*)(w + 3 * szPX + szS);
    float*  C   = (float*)(w + 3 * szPX + szS + szA);

    // 1) row softmax stats
    k_row_stats<<<dim3(BB * NN / 8), dim3(256), 0, stream>>>(x, P, XB, C);

    // 2) XBT = XB^T per batch
    k_transpose<<<dim3(NN / 64, DD / 64, BB), dim3(256), 0, stream>>>(XB, XBT);

    // 3) S = p @ x^T - c_j : grid (N/64, N/256, B)
    k_gemm1<<<dim3(NN / 64, NN / 256, BB), dim3(256), 0, stream>>>(P, XB, C, S);

    // 4) attn = softmax_j(S)
    k_attn_softmax<<<dim3(BB * NN / 8), dim3(256), 0, stream>>>(S, ATT);

    // 5) out = attn @ x : grid (N/64, D/256, B)
    k_gemm2<<<dim3(NN / 64, DD / 256, BB), dim3(256), 0, stream>>>(ATT, XBT, out);

    (void)in_sizes; (void)n_in; (void)out_size; (void)ws_size;
}
